// DAC_LSTMModel_CELL_47725676593631
// MI455X (gfx1250) — matrix operand order repaired and checked by reading, not compiled, not run on hardware
//
#include <hip/hip_runtime.h>
#include <hip/hip_bf16.h>

// ---------------------------------------------------------------------------
// Fused 2-layer LSTM scan for MI455X (gfx1250, wave32, WMMA).
// One workgroup owns a 16-row batch tile and runs the entire T=256 recurrence
// on-chip: all weights live in LDS as f16 (~237 KB of the 320 KB/WGP LDS),
// gates are computed with v_wmma_f32_16x16x32_f16, cell nonlinearities +
// dropout masking are done in-LDS. Only x tiles and masks stream from HBM.
// ---------------------------------------------------------------------------

typedef __attribute__((ext_vector_type(16))) _Float16 v16h;
typedef __attribute__((ext_vector_type(8)))  float    v8f;
typedef __attribute__((ext_vector_type(4)))  float    f32x4;

#define BSZ 512
#define TT  256
#define II  64
#define H0  128
#define H1  32
#define G0  (4*H0)   // 512
#define G1  (4*H1)   // 128
#define LD0 520      // padded row length (elems) for layer-0 WT + gates0
#define LD1 136      // padded row length for layer-1 WT + gates1

// LDS carve (bytes); all offsets 16B aligned.
#define OFF_WT_IH0 0u                       // [64 ][LD0] f16  66560
#define OFF_WT_HH0 66560u                   // [128][LD0] f16 133120
#define OFF_WT_IH1 199680u                  // [128][LD1] f16  34816
#define OFF_WT_HH1 234496u                  // [32 ][LD1] f16   8704
#define OFF_BIAS0  243200u                  // [512] f32        2048
#define OFF_BIAS1  245248u                  // [128] f32         512
#define OFF_AX     245760u                  // [16][64]  f16    2048
#define OFF_AH0    247808u                  // [16][128] f16    4096
#define OFF_AH1    251904u                  // [16][32]  f16    1024
#define OFF_G0     252928u                  // [16][LD0] f32   33280
#define OFF_G1     286208u                  // [16][LD1] f32    8704
#define OFF_C0     294912u                  // [16][128] f32    8192
#define OFF_C1     303104u                  // [16][32]  f32    2048
#define OFF_H1F    305152u                  // [16][32]  f32    2048
#define LDS_BYTES  307200u                  // 300 KB < 320 KB/WGP

__device__ __forceinline__ float sigf(float x) {
    return 1.0f / (1.0f + __expf(-x));
}
__device__ __forceinline__ float tanh_fast(float x) {
    x = fminf(fmaxf(x, -15.0f), 15.0f);
    float e = __expf(2.0f * x);
    return (e - 1.0f) / (e + 1.0f);
}

// A fragment (16x32 f16, M x K), ISA 7.12.2: lanes 0-15 hold K 0..7 & 16..23,
// lanes 16-31 hold K 8..15 & 24..31, for row M = lane&15.  Two 16B LDS loads.
__device__ __forceinline__ v16h ld_frag_a(const _Float16* base, int ld,
                                          int kbase, int lane) {
    const int m  = lane & 15;
    const int hl = lane >> 4;
    const _Float16* p = base + m * ld + kbase + hl * 8;
    union { f32x4 f[2]; v16h v; } u;
    u.f[0] = *reinterpret_cast<const f32x4*>(p);
    u.f[1] = *reinterpret_cast<const f32x4*>(p + 16);
    return u.v;
}

// B fragment (32x16 f16, K x N): lane l holds N = n0 + (l&15), K as in A:
// lanes 0-15 K 0..7 & 16..23, lanes 16-31 K 8..15 & 24..31 (strided by ld).
__device__ __forceinline__ v16h ld_frag_b(const _Float16* wt, int ld,
                                          int n0, int kbase, int lane) {
    const _Float16* p = wt + (size_t)(kbase + (lane >> 4) * 8) * ld + n0 + (lane & 15);
    v16h v;
    #pragma unroll
    for (int s = 0; s < 8; ++s) {
        v[s]     = p[(size_t)s * ld];
        v[s + 8] = p[(size_t)(16 + s) * ld];
    }
    return v;
}

__global__ __launch_bounds__(256, 1)
void lstm_fused_kernel(const float* __restrict__ xin,
                       const float* __restrict__ mh0, const float* __restrict__ mc0,
                       const float* __restrict__ mh1, const float* __restrict__ mc1,
                       const float* __restrict__ Wih0, const float* __restrict__ Whh0,
                       const float* __restrict__ bih0, const float* __restrict__ bhh0,
                       const float* __restrict__ Wih1, const float* __restrict__ Whh1,
                       const float* __restrict__ bih1, const float* __restrict__ bhh1,
                       const float* __restrict__ Wlin, const float* __restrict__ blin,
                       float* __restrict__ out) {
    extern __shared__ char smem[];
    _Float16* wt_ih0 = reinterpret_cast<_Float16*>(smem + OFF_WT_IH0);
    _Float16* wt_hh0 = reinterpret_cast<_Float16*>(smem + OFF_WT_HH0);
    _Float16* wt_ih1 = reinterpret_cast<_Float16*>(smem + OFF_WT_IH1);
    _Float16* wt_hh1 = reinterpret_cast<_Float16*>(smem + OFF_WT_HH1);
    float*    bias0  = reinterpret_cast<float*>(smem + OFF_BIAS0);
    float*    bias1  = reinterpret_cast<float*>(smem + OFF_BIAS1);
    _Float16* a_x    = reinterpret_cast<_Float16*>(smem + OFF_AX);
    _Float16* a_h0   = reinterpret_cast<_Float16*>(smem + OFF_AH0);
    _Float16* a_h1   = reinterpret_cast<_Float16*>(smem + OFF_AH1);
    float*    gates0 = reinterpret_cast<float*>(smem + OFF_G0);
    float*    gates1 = reinterpret_cast<float*>(smem + OFF_G1);
    float*    c0s    = reinterpret_cast<float*>(smem + OFF_C0);
    float*    c1s    = reinterpret_cast<float*>(smem + OFF_C1);
    float*    h1f    = reinterpret_cast<float*>(smem + OFF_H1F);

    const int tid   = threadIdx.x;
    const int lane  = tid & 31;
    const int wv    = tid >> 5;       // 8 waves
    const int hl    = lane >> 4;
    const int col   = lane & 15;
    const int bbase = blockIdx.x * 16;

    // ---- Phase 0: stage transposed f16 weights + fused biases in LDS ------
    for (int idx = tid; idx < G0 * II; idx += 256) {
        int g = idx >> 6, k = idx & 63;
        wt_ih0[k * LD0 + g] = (_Float16)Wih0[idx];
    }
    for (int idx = tid; idx < G0 * H0; idx += 256) {
        int g = idx >> 7, k = idx & 127;
        wt_hh0[k * LD0 + g] = (_Float16)Whh0[idx];
    }
    for (int idx = tid; idx < G1 * H0; idx += 256) {
        int g = idx >> 7, k = idx & 127;
        wt_ih1[k * LD1 + g] = (_Float16)Wih1[idx];
    }
    for (int idx = tid; idx < G1 * H1; idx += 256) {
        int g = idx >> 5, k = idx & 31;
        wt_hh1[k * LD1 + g] = (_Float16)Whh1[idx];
    }
    for (int g = tid; g < G0; g += 256) bias0[g] = bih0[g] + bhh0[g];
    for (int g = tid; g < G1; g += 256) bias1[g] = bih1[g] + bhh1[g];
    for (int idx = tid; idx < 16 * H0; idx += 256) { c0s[idx] = 0.f; a_h0[idx] = (_Float16)0.f; }
    for (int idx = tid; idx < 16 * H1; idx += 256) { c1s[idx] = 0.f; a_h1[idx] = (_Float16)0.f; }
    __syncthreads();

    // ---- Recurrence over T, entirely on-chip ------------------------------
    for (int t = 0; t < TT; ++t) {
        // Step 1: stage x[t] tile as f16 A-operand
        for (int idx = tid; idx < 16 * II; idx += 256) {
            int m = idx >> 6, i = idx & 63;
            a_x[idx] = (_Float16)xin[((size_t)(bbase + m) * TT + t) * II + i];
        }
        __syncthreads();

        // Step 2: gates0[16 x 512] = x@Wih0^T + h0@Whh0^T + b   (WMMA)
        for (int j = 0; j < 4; ++j) {
            const int n0 = (wv * 4 + j) * 16;
            float bv = bias0[n0 + col];
            v8f acc;
            #pragma unroll
            for (int r = 0; r < 8; ++r) acc[r] = bv;
            #pragma unroll
            for (int ks = 0; ks < 2; ++ks) {   // K = 64 from x
                v16h a = ld_frag_a(a_x, II, ks * 32, lane);
                v16h b = ld_frag_b(wt_ih0, LD0, n0, ks * 32, lane);
                acc = __builtin_amdgcn_wmma_f32_16x16x32_f16(
                          false, a, false, b, (short)0, acc, false, false);
            }
            #pragma unroll
            for (int ks = 0; ks < 4; ++ks) {   // K = 128 from h0
                v16h a = ld_frag_a(a_h0, H0, ks * 32, lane);
                v16h b = ld_frag_b(wt_hh0, LD0, n0, ks * 32, lane);
                acc = __builtin_amdgcn_wmma_f32_16x16x32_f16(
                          false, a, false, b, (short)0, acc, false, false);
            }
            #pragma unroll
            for (int r = 0; r < 8; ++r)
                gates0[(r + hl * 8) * LD0 + n0 + col] = acc[r];
        }
        __syncthreads();

        // Step 3: layer-0 cell + dropout masks
        #pragma unroll
        for (int e = 0; e < 8; ++e) {
            int idx = tid + e * 256;          // 16*128 elems
            int m = idx >> 7, j = idx & 127;
            const float* gr = gates0 + m * LD0;
            float ig = gr[j], fg = gr[H0 + j], gg = gr[2 * H0 + j], og = gr[3 * H0 + j];
            float cold = c0s[m * H0 + j];
            float cn = sigf(fg) * cold + sigf(ig) * tanh_fast(gg);
            size_t mo = ((size_t)t * BSZ + (bbase + m)) * H0 + j;
            c0s[m * H0 + j] = cn * mc0[mo];
            float h = sigf(og) * tanh_fast(cn) * mh0[mo];
            a_h0[m * H0 + j] = (_Float16)h;
            if (t + 1 < TT) {                 // prefetch next step's masks
                __builtin_prefetch(&mc0[mo + (size_t)BSZ * H0], 0, 0);
                __builtin_prefetch(&mh0[mo + (size_t)BSZ * H0], 0, 0);
            }
        }
        __syncthreads();

        // Step 4: gates1[16 x 128] = h0@Wih1^T + h1@Whh1^T + b   (WMMA)
        {
            const int n0 = wv * 16;
            float bv = bias1[n0 + col];
            v8f acc;
            #pragma unroll
            for (int r = 0; r < 8; ++r) acc[r] = bv;
            #pragma unroll
            for (int ks = 0; ks < 4; ++ks) {  // K = 128 from h0
                v16h a = ld_frag_a(a_h0, H0, ks * 32, lane);
                v16h b = ld_frag_b(wt_ih1, LD1, n0, ks * 32, lane);
                acc = __builtin_amdgcn_wmma_f32_16x16x32_f16(
                          false, a, false, b, (short)0, acc, false, false);
            }
            {                                 // K = 32 from h1
                v16h a = ld_frag_a(a_h1, H1, 0, lane);
                v16h b = ld_frag_b(wt_hh1, LD1, n0, 0, lane);
                acc = __builtin_amdgcn_wmma_f32_16x16x32_f16(
                          false, a, false, b, (short)0, acc, false, false);
            }
            #pragma unroll
            for (int r = 0; r < 8; ++r)
                gates1[(r + hl * 8) * LD1 + n0 + col] = acc[r];
        }
        __syncthreads();

        // Step 5: layer-1 cell + dropout masks
        #pragma unroll
        for (int e = 0; e < 2; ++e) {
            int idx = tid + e * 256;          // 16*32 elems
            int m = idx >> 5, j = idx & 31;
            const float* gr = gates1 + m * LD1;
            float ig = gr[j], fg = gr[H1 + j], gg = gr[2 * H1 + j], og = gr[3 * H1 + j];
            float cold = c1s[m * H1 + j];
            float cn = sigf(fg) * cold + sigf(ig) * tanh_fast(gg);
            size_t mo = ((size_t)t * BSZ + (bbase + m)) * H1 + j;
            c1s[m * H1 + j] = cn * mc1[mo];
            float h = sigf(og) * tanh_fast(cn) * mh1[mo];
            a_h1[m * H1 + j] = (_Float16)h;
            h1f[m * H1 + j] = h;
            if (t + 1 < TT) {
                __builtin_prefetch(&mc1[mo + (size_t)BSZ * H1], 0, 0);
                __builtin_prefetch(&mh1[mo + (size_t)BSZ * H1], 0, 0);
            }
        }
        __syncthreads();
    }

    // ---- Final linear head: out[b] = h1 . W_lin + b_lin -------------------
    if (tid < 16) {
        int m = tid;
        float acc = blin[0];
        #pragma unroll
        for (int j = 0; j < H1; ++j) acc += h1f[m * H1 + j] * Wlin[j];
        out[bbase + m] = acc;
    }
}

extern "C" void kernel_launch(void* const* d_in, const int* in_sizes, int n_in,
                              void* d_out, int out_size, void* d_ws, size_t ws_size,
                              hipStream_t stream) {
    (void)in_sizes; (void)n_in; (void)d_ws; (void)ws_size; (void)out_size;

    const float* xin  = (const float*)d_in[0];
    const float* mh0  = (const float*)d_in[1];
    const float* mc0  = (const float*)d_in[2];
    const float* mh1  = (const float*)d_in[3];
    const float* mc1  = (const float*)d_in[4];
    const float* Wih0 = (const float*)d_in[5];
    const float* Whh0 = (const float*)d_in[6];
    const float* bih0 = (const float*)d_in[7];
    const float* bhh0 = (const float*)d_in[8];
    const float* Wih1 = (const float*)d_in[9];
    const float* Whh1 = (const float*)d_in[10];
    const float* bih1 = (const float*)d_in[11];
    const float* bhh1 = (const float*)d_in[12];
    const float* Wlin = (const float*)d_in[13];
    const float* blin = (const float*)d_in[14];
    float* out = (float*)d_out;

    // Allow >64KB dynamic LDS (gfx1250 WGP has 320 KB).
    (void)hipFuncSetAttribute(reinterpret_cast<const void*>(lstm_fused_kernel),
                              hipFuncAttributeMaxDynamicSharedMemorySize,
                              (int)LDS_BYTES);

    lstm_fused_kernel<<<dim3(BSZ / 16), dim3(256), LDS_BYTES, stream>>>(
        xin, mh0, mc0, mh1, mc1,
        Wih0, Whh0, bih0, bhh0,
        Wih1, Whh1, bih1, bhh1,
        Wlin, blin, out);
}
